// ContextualTransformer_65146063946464
// MI455X (gfx1250) — compile-verified, matrix operand order checked by reading, not run on hardware
//
#include <hip/hip_runtime.h>
#include <hip/hip_bf16.h>
#include <stdint.h>

// ---------------------------------------------------------------------------
// Model constants (from the reference)
// ---------------------------------------------------------------------------
#define BATCH   2
#define SEQT    1536
#define DMOT    256
#define DMODEL  512
#define NHEADS  8
#define DHEAD   64
#define DFF     2048
#define NLAYERS 6
#define MROWS   (BATCH * SEQT)      // 3072
#define DIN_PAD 288                 // 262 padded up to a multiple of 32
#define DIN_REAL 262
#define PERIOD  15                  // FPS // 2
#define THR     128                 // ceil(1525/2/6) -> ksz=257, thr=128
#define GROW    128                 // cf dilation per layer (ksz//2)
#define CF_LO0  9                   // context_frames - 1
#define NEGBIG  (-1.0e9f)

typedef __attribute__((ext_vector_type(16))) __bf16 v16bf;
typedef __attribute__((ext_vector_type(8)))  float  v8f;
typedef __attribute__((ext_vector_type(4)))  unsigned v4u;
typedef __attribute__((ext_vector_type(8)))  unsigned v8u;

union Frag16 {
    unsigned int u[8];
    v16bf        v;
};

__device__ __forceinline__ unsigned short f2bf(float f) {
    unsigned int u = __float_as_uint(f);
    u += 0x7FFFu + ((u >> 16) & 1u);        // round-to-nearest-even
    return (unsigned short)(u >> 16);
}
__device__ __forceinline__ unsigned int pack2(unsigned short lo, unsigned short hi) {
    return (unsigned int)lo | ((unsigned int)hi << 16);
}
// K-offset base for 16-bit A/B fragment register v (ISA 7.12.2 layout)
__device__ __forceinline__ int kbase(int v, int lane) {
    return ((v & 3) << 1) + ((v >> 2) << 4) + ((lane & 16) ? 8 : 0);
}

// CDNA5 async global->LDS copy: 16 bytes per lane, tracked by ASYNCcnt.
__device__ __forceinline__ void async_ld16(const void* lds_dst, const void* gsrc) {
    unsigned lds_off = (unsigned)(uintptr_t)lds_dst;     // addrspace(3) offset
    unsigned long long ga = (unsigned long long)(uintptr_t)gsrc;
    asm volatile("global_load_async_to_lds_b128 %0, %1, off"
                 :: "v"(lds_off), "v"(ga) : "memory");
}
__device__ __forceinline__ void wait_async0() {
    asm volatile("s_wait_asynccnt 0x0" ::: "memory");
}

// ---------------------------------------------------------------------------
// Tensor Data Mover: 2D tile (tile_d0 elems x tile_d1 rows of 2-byte data)
// global -> LDS, one instruction per tile, tracked by TENSORcnt (ISA ch.8).
// ---------------------------------------------------------------------------
__device__ __forceinline__ v8u tdm_desc2d(unsigned tile_d0, unsigned tile_d1,
                                          unsigned ten_d0, unsigned ten_d1,
                                          unsigned stride0)
{
    v8u g1;
    g1[0] = 1u << 16;                                   // data_size = 2 bytes
    g1[1] = (ten_d0 & 0xFFFFu) << 16;                   // tensor_dim0[15:0]
    g1[2] = (ten_d0 >> 16) | ((ten_d1 & 0xFFFFu) << 16);// dim0 hi | dim1 lo
    g1[3] = (ten_d1 >> 16) | (tile_d0 << 16);           // dim1 hi | tile_dim0
    g1[4] = tile_d1 & 0xFFFFu;                          // tile_dim1 (tile_dim2=0)
    g1[5] = stride0;                                    // tensor_dim0_stride lo32
    g1[6] = 0u;                                         // stride hi | dim1_stride lo
    g1[7] = 0u;
    return g1;
}
__device__ __forceinline__ void tdm_issue(unsigned lds_off, unsigned long long ga, v8u g1)
{
    v4u g0;
    g0[0] = 1u;                                         // count=1, user descriptor
    g0[1] = lds_off;                                    // LDS byte address
    g0[2] = (unsigned)ga;                               // global addr [31:0]
    g0[3] = (unsigned)(ga >> 32) | (2u << 30);          // addr[56:32] | type=2
    v4u gz = {0u, 0u, 0u, 0u};
    asm volatile("tensor_load_to_lds %0, %1, %2, %3"
                 :: "s"(g0), "s"(g1), "s"(gz), "s"(gz) : "memory");
}

// ---------------------------------------------------------------------------
// Weight convert: Wt[n*Kpad + k] = bf16(W[k*N + n]), zero beyond Kreal.
// ---------------------------------------------------------------------------
__global__ void wt_kernel(const float* __restrict__ W, unsigned short* __restrict__ Wt,
                          int Kpad, int Kreal, int N)
{
    int idx = blockIdx.x * blockDim.x + threadIdx.x;
    if (idx >= N * Kpad) return;
    int n = idx / Kpad, k = idx % Kpad;
    float v = (k < Kreal) ? W[k * N + n] : 0.0f;
    Wt[idx] = f2bf(v);
}

// ---------------------------------------------------------------------------
// GEMM: act(A[M,K]bf16 @ W[K,N] + bias (+ res fp32)) -> Cf fp32 / Cb bf16
// Wt is the transposed bf16 weight [N][K]. Tile 128x64 per WG (4 waves, each
// 32x64 -> 8 WMMAs per 32-K step). Tiles staged via TDM (one tensor_load per
// tile, issued by wave 0). ACT: 0 = none, 1 = leaky(0.01), 2 = relu.
// ---------------------------------------------------------------------------
template <int ACT, bool HAS_RES, bool HAS_CF, bool HAS_CB>
__global__ __launch_bounds__(128) void gemm_tdm_kernel(
    const unsigned short* __restrict__ A, const unsigned short* __restrict__ Wt,
    const float* __restrict__ bias, const float* __restrict__ res,
    float* __restrict__ Cf, unsigned short* __restrict__ Cb,
    int M, int N, int K)
{
    __shared__ __align__(16) unsigned short As[128 * 32];  // [m][k]
    __shared__ __align__(16) unsigned short Bs[64 * 32];   // [n][k]

    const int tid  = threadIdx.x;
    const int lane = tid & 31;
    const int wave = tid >> 5;
    const int m0   = blockIdx.y * 128;
    const int n0   = blockIdx.x * 64;

    v8f acc[8];
    v8f zero = {0.f, 0.f, 0.f, 0.f, 0.f, 0.f, 0.f, 0.f};
#pragma unroll
    for (int i = 0; i < 8; ++i) acc[i] = zero;

    const int mbase = wave * 32;
    const int m16   = lane & 15;

    const v8u dA = tdm_desc2d(32, 128, (unsigned)K, (unsigned)M, (unsigned)K);
    const v8u dB = tdm_desc2d(32, 64,  (unsigned)K, (unsigned)N, (unsigned)K);
    const unsigned ldsA = (unsigned)(uintptr_t)&As[0];
    const unsigned ldsB = (unsigned)(uintptr_t)&Bs[0];

    for (int kt = 0; kt < K; kt += 32) {
        if (wave == 0) {
            tdm_issue(ldsA, (unsigned long long)(uintptr_t)(A + (size_t)m0 * K + kt), dA);
            tdm_issue(ldsB, (unsigned long long)(uintptr_t)(Wt + (size_t)n0 * K + kt), dB);
            __builtin_amdgcn_s_wait_tensorcnt(0);
        }
        __syncthreads();

        Frag16 fa0, fa1;
#pragma unroll
        for (int v = 0; v < 8; ++v) {
            int kb = kbase(v, lane);
            fa0.u[v] = *(const unsigned int*)&As[(mbase + m16) * 32 + kb];
            fa1.u[v] = *(const unsigned int*)&As[(mbase + 16 + m16) * 32 + kb];
        }
#pragma unroll
        for (int nt = 0; nt < 4; ++nt) {
            Frag16 fb;
            int nn = nt * 16 + m16;
#pragma unroll
            for (int v = 0; v < 8; ++v)
                fb.u[v] = *(const unsigned int*)&Bs[nn * 32 + kbase(v, lane)];
            acc[nt]     = __builtin_amdgcn_wmma_f32_16x16x32_bf16(
                false, fa0.v, false, fb.v, (short)0, acc[nt], false, false);
            acc[4 + nt] = __builtin_amdgcn_wmma_f32_16x16x32_bf16(
                false, fa1.v, false, fb.v, (short)0, acc[4 + nt], false, false);
        }
        __syncthreads();
    }

    const int hi8 = (lane & 16) ? 8 : 0;
    const int c16 = lane & 15;
#pragma unroll
    for (int half = 0; half < 2; ++half) {
#pragma unroll
        for (int nt = 0; nt < 4; ++nt) {
            int gn = n0 + nt * 16 + c16;
            float bv = bias[gn];
#pragma unroll
            for (int r = 0; r < 8; ++r) {
                int grow = m0 + mbase + half * 16 + r + hi8;
                float v = acc[half * 4 + nt][r] + bv;
                if (HAS_RES) v += res[(size_t)grow * N + gn];
                if (ACT == 1)      v = (v >= 0.f) ? v : 0.01f * v;
                else if (ACT == 2) v = fmaxf(v, 0.f);
                if (HAS_CF) Cf[(size_t)grow * N + gn] = v;
                if (HAS_CB) Cb[(size_t)grow * N + gn] = f2bf(v);
            }
        }
    }
}

// ---------------------------------------------------------------------------
// Flash-style attention, all-bf16 I/O. Grid: B*H*(T/64) WGs of 128 threads.
// Each wave owns 16 queries x 64 head-dim; streams 32-key tiles (async-staged).
// Bias: -(|dq-dk|//15); mask: dist>128 or key not constrained at this layer.
// ---------------------------------------------------------------------------
__global__ __launch_bounds__(128) void attn_kernel(
    const unsigned short* __restrict__ Q, const unsigned short* __restrict__ Kb,
    const unsigned short* __restrict__ V, unsigned short* __restrict__ O, int layer)
{
    __shared__ __align__(16) unsigned short Ks[32 * 64];     // [key][d]
    __shared__ __align__(16) unsigned short Vs[32 * 64];     // [key][d]
    __shared__ __align__(16) unsigned short Ps[4][16 * 32];  // per-wave P [m][key]

    const int tid  = threadIdx.x;
    const int lane = tid & 31;
    const int wave = tid >> 5;
    const int wg   = blockIdx.x;
    const int qblk = wg % (SEQT / 64);
    const int h    = (wg / (SEQT / 64)) % NHEADS;
    const int b    = wg / ((SEQT / 64) * NHEADS);
    const int rowbase = b * SEQT;
    const int q0   = qblk * 64 + wave * 16;
    const int m    = lane & 15;
    const int hi8  = (lane & 16) ? 8 : 0;

    // Q fragments: direct bf16 pair loads from global
    Frag16 qa0, qa1;
    {
        const unsigned short* qrow = Q + (size_t)(rowbase + q0 + m) * DMODEL + h * DHEAD;
#pragma unroll
        for (int v = 0; v < 8; ++v) {
            int kb = kbase(v, lane);
            qa0.u[v] = *(const unsigned int*)&qrow[kb];
            qa1.u[v] = *(const unsigned int*)&qrow[32 + kb];
        }
    }

    v8f zero = {0.f, 0.f, 0.f, 0.f, 0.f, 0.f, 0.f, 0.f};
    v8f o[4];
#pragma unroll
    for (int i = 0; i < 4; ++i) o[i] = zero;
    float rm[8], rs[8];
#pragma unroll
    for (int r = 0; r < 8; ++r) { rm[r] = -3.0e38f; rs[r] = 0.f; }

    const int cf_lo = CF_LO0 + GROW * layer;        // t <= cf_lo  is constrained
    const int cf_hi = (SEQT - 1) - GROW * layer;    // t >= cf_hi  is constrained
    const float scale = 0.125f;                     // 1/sqrt(64)
    const int n = lane & 15;
    unsigned short* Pw = &Ps[wave][0];

    for (int kt0 = 0; kt0 < SEQT; kt0 += 32) {
        // async-stage K and V tiles (each 32 keys x 128B)
#pragma unroll
        for (int j = 0; j < 2; ++j) {
            int lin = j * 128 + tid;
            int key = lin >> 3, oc = lin & 7;
            size_t g = (size_t)(rowbase + kt0 + key) * DMODEL + h * DHEAD + oc * 8;
            async_ld16(&Ks[key * 64 + oc * 8], Kb + g);
            async_ld16(&Vs[key * 64 + oc * 8], V + g);
        }
        wait_async0();
        __syncthreads();

        // S = Q @ K^T for two 16-key subtiles
        v8f s0 = zero, s1 = zero;
        Frag16 fb;
#pragma unroll
        for (int v = 0; v < 8; ++v)
            fb.u[v] = *(const unsigned int*)&Ks[n * 64 + kbase(v, lane)];
        s0 = __builtin_amdgcn_wmma_f32_16x16x32_bf16(false, qa0.v, false, fb.v, (short)0, s0, false, false);
#pragma unroll
        for (int v = 0; v < 8; ++v)
            fb.u[v] = *(const unsigned int*)&Ks[n * 64 + 32 + kbase(v, lane)];
        s0 = __builtin_amdgcn_wmma_f32_16x16x32_bf16(false, qa1.v, false, fb.v, (short)0, s0, false, false);
#pragma unroll
        for (int v = 0; v < 8; ++v)
            fb.u[v] = *(const unsigned int*)&Ks[(n + 16) * 64 + kbase(v, lane)];
        s1 = __builtin_amdgcn_wmma_f32_16x16x32_bf16(false, qa0.v, false, fb.v, (short)0, s1, false, false);
#pragma unroll
        for (int v = 0; v < 8; ++v)
            fb.u[v] = *(const unsigned int*)&Ks[(n + 16) * 64 + 32 + kbase(v, lane)];
        s1 = __builtin_amdgcn_wmma_f32_16x16x32_bf16(false, qa1.v, false, fb.v, (short)0, s1, false, false);

        // bias + mask + online softmax (row reductions within 16-lane halves)
        const int tk0 = kt0 + n, tk1 = tk0 + 16;
        const bool cf0 = (tk0 <= cf_lo) || (tk0 >= cf_hi);
        const bool cf1 = (tk1 <= cf_lo) || (tk1 >= cf_hi);
#pragma unroll
        for (int r = 0; r < 8; ++r) {
            int tq = q0 + r + hi8;
            int d0 = (tq > tk0) ? (tq - tk0) : (tk0 - tq);
            int d1 = (tq > tk1) ? (tq - tk1) : (tk1 - tq);
            float v0 = (d0 > THR || !cf0) ? NEGBIG : (s0[r] * scale - (float)(d0 / PERIOD));
            float v1 = (d1 > THR || !cf1) ? NEGBIG : (s1[r] * scale - (float)(d1 / PERIOD));
            float mx = fmaxf(v0, v1);
            mx = fmaxf(mx, __shfl_xor(mx, 1));
            mx = fmaxf(mx, __shfl_xor(mx, 2));
            mx = fmaxf(mx, __shfl_xor(mx, 4));
            mx = fmaxf(mx, __shfl_xor(mx, 8));
            float nm  = fmaxf(rm[r], mx);
            float fac = __expf(rm[r] - nm);
            rm[r] = nm;
            float p0 = __expf(v0 - nm), p1 = __expf(v1 - nm);
            float ps = p0 + p1;
            ps += __shfl_xor(ps, 1);
            ps += __shfl_xor(ps, 2);
            ps += __shfl_xor(ps, 4);
            ps += __shfl_xor(ps, 8);
            rs[r] = rs[r] * fac + ps;
            o[0][r] *= fac; o[1][r] *= fac; o[2][r] *= fac; o[3][r] *= fac;
            Pw[(r + hi8) * 32 + n]      = f2bf(p0);
            Pw[(r + hi8) * 32 + 16 + n] = f2bf(p1);
        }
        // same-wave LDS ops stay in order (DScnt)

        // O += P @ V  (V transposed during fragment build: pair gathers)
        Frag16 pa;
#pragma unroll
        for (int v = 0; v < 8; ++v)
            pa.u[v] = *(const unsigned int*)&Pw[m * 32 + kbase(v, lane)];
#pragma unroll
        for (int nt = 0; nt < 4; ++nt) {
            Frag16 vb;
            int dcol = nt * 16 + n;
#pragma unroll
            for (int v = 0; v < 8; ++v) {
                int kv = kbase(v, lane);
                vb.u[v] = pack2(Vs[kv * 64 + dcol], Vs[(kv + 1) * 64 + dcol]);
            }
            o[nt] = __builtin_amdgcn_wmma_f32_16x16x32_bf16(
                false, pa.v, false, vb.v, (short)0, o[nt], false, false);
        }
        __syncthreads();
    }

#pragma unroll
    for (int nt = 0; nt < 4; ++nt)
#pragma unroll
        for (int r = 0; r < 8; ++r) {
            int grow = rowbase + q0 + r + hi8;
            O[(size_t)grow * DMODEL + h * DHEAD + nt * 16 + n] = f2bf(o[nt][r] / rs[r]);
        }
}

// ---------------------------------------------------------------------------
// LayerNorm: one 128-thread block per row of 512; fp32 in, bf16 out.
// ---------------------------------------------------------------------------
__global__ __launch_bounds__(128) void ln_kernel(
    const float* __restrict__ X, const float* __restrict__ g,
    const float* __restrict__ bta, unsigned short* __restrict__ Y)
{
    __shared__ float red[128];
    const int row = blockIdx.x;
    const int tid = threadIdx.x;
    const float* x = X + (size_t)row * DMODEL;
    float lv[4];
    float s = 0.f;
#pragma unroll
    for (int i = 0; i < 4; ++i) { lv[i] = x[tid + i * 128]; s += lv[i]; }
    red[tid] = s; __syncthreads();
    for (int off = 64; off > 0; off >>= 1) {
        if (tid < off) red[tid] += red[tid + off];
        __syncthreads();
    }
    float mean = red[0] * (1.0f / DMODEL);
    __syncthreads();
    float vs = 0.f;
#pragma unroll
    for (int i = 0; i < 4; ++i) { float d = lv[i] - mean; vs += d * d; }
    red[tid] = vs; __syncthreads();
    for (int off = 64; off > 0; off >>= 1) {
        if (tid < off) red[tid] += red[tid + off];
        __syncthreads();
    }
    float rstd = rsqrtf(red[0] * (1.0f / DMODEL) + 1e-5f);
#pragma unroll
    for (int i = 0; i < 4; ++i) {
        int c = tid + i * 128;
        Y[(size_t)row * DMODEL + c] = f2bf((lv[i] - mean) * rstd * g[c] + bta[c]);
    }
}

// ---------------------------------------------------------------------------
// Pack encoder input (bf16): [motion*mask | traj | mask | zero-pad] -> [3072,288]
// ---------------------------------------------------------------------------
__global__ void build_input_kernel(const float* __restrict__ motion,
                                   const float* __restrict__ traj,
                                   unsigned short* __restrict__ X0)
{
    int idx = blockIdx.x * blockDim.x + threadIdx.x;
    if (idx >= MROWS * DIN_PAD) return;
    int c   = idx % DIN_PAD;
    int row = idx / DIN_PAD;
    int t   = row % SEQT;
    float mk = (t < 10 || t == SEQT - 1) ? 1.f : 0.f;
    float v;
    if (c < DMOT)        v = motion[row * DMOT + c] * mk;
    else if (c < 261)    v = traj[row * 5 + (c - DMOT)];
    else if (c == 261)   v = mk;
    else                 v = 0.f;
    X0[idx] = f2bf(v);
}

// ---------------------------------------------------------------------------
// Periodic sinusoidal positional embedding add (pos = t mod 15), fp32.
// ---------------------------------------------------------------------------
__global__ void pe_add_kernel(float* __restrict__ X)
{
    int idx = blockIdx.x * blockDim.x + threadIdx.x;
    if (idx >= MROWS * DMODEL) return;
    int c   = idx & (DMODEL - 1);
    int row = idx >> 9;
    int t   = row % SEQT;
    float pos = (float)(t % PERIOD);
    int j = (c < DMODEL / 2) ? c : (c - DMODEL / 2);
    float div = __expf(-9.210340371976184f * (float)(2 * j) * (1.0f / DMODEL));
    float ang = pos * div;
    X[idx] += (c < DMODEL / 2) ? __sinf(ang) : __cosf(ang);
}

// ---------------------------------------------------------------------------
// Final blend: out = motion*mask + dec*(1-mask); also emit the mask output.
// ---------------------------------------------------------------------------
__global__ void blend_kernel(const float* __restrict__ motion,
                             const float* __restrict__ dec,
                             float* __restrict__ out,
                             float* __restrict__ outmask)
{
    int idx = blockIdx.x * blockDim.x + threadIdx.x;
    if (idx < MROWS * DMOT) {
        int row = idx >> 8;
        int t   = row % SEQT;
        float mk = (t < 10 || t == SEQT - 1) ? 1.f : 0.f;
        out[idx] = mk * motion[idx] + (1.f - mk) * dec[idx];
    }
    if (idx < MROWS) {
        int t = idx % SEQT;
        outmask[idx] = (t < 10 || t == SEQT - 1) ? 1.f : 0.f;
    }
}

// ---------------------------------------------------------------------------
// Orchestration
// ---------------------------------------------------------------------------
extern "C" void kernel_launch(void* const* d_in, const int* in_sizes, int n_in,
                              void* d_out, int out_size, void* d_ws, size_t ws_size,
                              hipStream_t stream)
{
    const float* motion = (const float*)d_in[0];
    const float* traj   = (const float*)d_in[1];
    const float* me_w1  = (const float*)d_in[2];
    const float* me_b1  = (const float*)d_in[3];
    const float* me_w2  = (const float*)d_in[4];
    const float* me_b2  = (const float*)d_in[5];
    const float* wq     = (const float*)d_in[6];
    const float* bq     = (const float*)d_in[7];
    const float* wk     = (const float*)d_in[8];
    const float* bk     = (const float*)d_in[9];
    const float* wv     = (const float*)d_in[10];
    const float* bv     = (const float*)d_in[11];
    const float* wo     = (const float*)d_in[12];
    const float* bo     = (const float*)d_in[13];
    const float* ln1_g  = (const float*)d_in[14];
    const float* ln1_b  = (const float*)d_in[15];
    const float* ffn_w1 = (const float*)d_in[16];
    const float* ffn_b1 = (const float*)d_in[17];
    const float* ffn_w2 = (const float*)d_in[18];
    const float* ffn_b2 = (const float*)d_in[19];
    const float* ln2_g  = (const float*)d_in[20];
    const float* ln2_b  = (const float*)d_in[21];
    const float* fln_g  = (const float*)d_in[22];
    const float* fln_b  = (const float*)d_in[23];
    const float* dec_w1 = (const float*)d_in[24];
    const float* dec_b1 = (const float*)d_in[25];
    const float* dec_w2 = (const float*)d_in[26];
    const float* dec_b2 = (const float*)d_in[27];

    float* out = (float*)d_out;

    // ---- workspace layout ----
    float* buf_x = (float*)d_ws;                         // [3072,512] fp32 residual
    float* buf_d = buf_x + (size_t)MROWS * DMODEL;       // [3072,256] fp32 decoder out
    unsigned short* us = (unsigned short*)(buf_d + (size_t)MROWS * DMOT);
    unsigned short* bX0 = us; us += (size_t)MROWS * DIN_PAD;
    unsigned short* bH  = us; us += (size_t)MROWS * DMODEL;
    unsigned short* bQ  = us; us += (size_t)MROWS * DMODEL;
    unsigned short* bK  = us; us += (size_t)MROWS * DMODEL;
    unsigned short* bV  = us; us += (size_t)MROWS * DMODEL;
    unsigned short* bA  = us; us += (size_t)MROWS * DMODEL;
    unsigned short* bFF = us; us += (size_t)MROWS * DFF;
    unsigned short* wt_me1 = us; us += (size_t)DMODEL * DIN_PAD;
    unsigned short* wt_me2 = us; us += (size_t)DMODEL * DMODEL;
    unsigned short* wt_q   = us; us += (size_t)NLAYERS * DMODEL * DMODEL;
    unsigned short* wt_k   = us; us += (size_t)NLAYERS * DMODEL * DMODEL;
    unsigned short* wt_v   = us; us += (size_t)NLAYERS * DMODEL * DMODEL;
    unsigned short* wt_o   = us; us += (size_t)NLAYERS * DMODEL * DMODEL;
    unsigned short* wt_f1  = us; us += (size_t)NLAYERS * DFF * DMODEL;
    unsigned short* wt_f2  = us; us += (size_t)NLAYERS * DMODEL * DFF;
    unsigned short* wt_d1  = us; us += (size_t)DMODEL * DMODEL;
    unsigned short* wt_d2  = us; us += (size_t)DMOT * DMODEL;

    const dim3 blk128(128);
    const dim3 g512(DMODEL / 64, MROWS / 128);           // (8, 24)
    const dim3 g2048(DFF / 64, MROWS / 128);             // (32, 24)
    const dim3 g256(DMOT / 64, MROWS / 128);             // (4, 24)

    // ---- one-time (per call) weight conversion to transposed bf16 ----
#define CVT(W, WT, KP, KR, NN) \
    wt_kernel<<<((NN) * (KP) + 255) / 256, 256, 0, stream>>>((W), (WT), (KP), (KR), (NN))
    CVT(me_w1, wt_me1, DIN_PAD, DIN_REAL, DMODEL);
    CVT(me_w2, wt_me2, DMODEL, DMODEL, DMODEL);
    for (int l = 0; l < NLAYERS; ++l) {
        const size_t w5 = (size_t)l * DMODEL * DMODEL;
        CVT(wq + w5, wt_q + w5, DMODEL, DMODEL, DMODEL);
        CVT(wk + w5, wt_k + w5, DMODEL, DMODEL, DMODEL);
        CVT(wv + w5, wt_v + w5, DMODEL, DMODEL, DMODEL);
        CVT(wo + w5, wt_o + w5, DMODEL, DMODEL, DMODEL);
        CVT(ffn_w1 + (size_t)l * DMODEL * DFF, wt_f1 + (size_t)l * DFF * DMODEL,
            DMODEL, DMODEL, DFF);
        CVT(ffn_w2 + (size_t)l * DFF * DMODEL, wt_f2 + (size_t)l * DMODEL * DFF,
            DFF, DFF, DMODEL);
    }
    CVT(dec_w1, wt_d1, DMODEL, DMODEL, DMODEL);
    CVT(dec_w2, wt_d2, DMODEL, DMODEL, DMOT);
#undef CVT

    // ---- motion encoder ----
    build_input_kernel<<<(MROWS * DIN_PAD + 255) / 256, 256, 0, stream>>>(motion, traj, bX0);
    gemm_tdm_kernel<1, false, false, true><<<g512, blk128, 0, stream>>>(
        bX0, wt_me1, me_b1, nullptr, nullptr, bH, MROWS, DMODEL, DIN_PAD);
    gemm_tdm_kernel<1, false, true, false><<<g512, blk128, 0, stream>>>(
        bH, wt_me2, me_b2, nullptr, buf_x, nullptr, MROWS, DMODEL, DMODEL);
    pe_add_kernel<<<(MROWS * DMODEL + 255) / 256, 256, 0, stream>>>(buf_x);

    // ---- transformer layers ----
    for (int i = 0; i < NLAYERS; ++i) {
        const size_t w5 = (size_t)i * DMODEL * DMODEL;
        const size_t b5 = (size_t)i * DMODEL;
        ln_kernel<<<MROWS, blk128, 0, stream>>>(buf_x, ln1_g + b5, ln1_b + b5, bH);
        gemm_tdm_kernel<0, false, false, true><<<g512, blk128, 0, stream>>>(
            bH, wt_q + w5, bq + b5, nullptr, nullptr, bQ, MROWS, DMODEL, DMODEL);
        gemm_tdm_kernel<0, false, false, true><<<g512, blk128, 0, stream>>>(
            bH, wt_k + w5, bk + b5, nullptr, nullptr, bK, MROWS, DMODEL, DMODEL);
        gemm_tdm_kernel<0, false, false, true><<<g512, blk128, 0, stream>>>(
            bH, wt_v + w5, bv + b5, nullptr, nullptr, bV, MROWS, DMODEL, DMODEL);
        attn_kernel<<<BATCH * NHEADS * (SEQT / 64), blk128, 0, stream>>>(bQ, bK, bV, bA, i);
        gemm_tdm_kernel<0, true, true, false><<<g512, blk128, 0, stream>>>(
            bA, wt_o + w5, bo + b5, buf_x, buf_x, nullptr, MROWS, DMODEL, DMODEL);
        ln_kernel<<<MROWS, blk128, 0, stream>>>(buf_x, ln2_g + b5, ln2_b + b5, bH);
        gemm_tdm_kernel<2, false, false, true><<<g2048, blk128, 0, stream>>>(
            bH, wt_f1 + (size_t)i * DFF * DMODEL, ffn_b1 + (size_t)i * DFF, nullptr,
            nullptr, bFF, MROWS, DFF, DMODEL);
        gemm_tdm_kernel<0, true, true, false><<<g512, blk128, 0, stream>>>(
            bFF, wt_f2 + (size_t)i * DMODEL * DFF, ffn_b2 + b5, buf_x,
            buf_x, nullptr, MROWS, DMODEL, DFF);
    }

    // ---- decoder + blend ----
    ln_kernel<<<MROWS, blk128, 0, stream>>>(buf_x, fln_g, fln_b, bH);
    gemm_tdm_kernel<1, false, false, true><<<g512, blk128, 0, stream>>>(
        bH, wt_d1, dec_b1, nullptr, nullptr, bA, MROWS, DMODEL, DMODEL);
    gemm_tdm_kernel<0, false, true, false><<<g256, blk128, 0, stream>>>(
        bA, wt_d2, dec_b2, nullptr, buf_d, nullptr, MROWS, DMOT, DMODEL);
    blend_kernel<<<(MROWS * DMOT + 255) / 256, 256, 0, stream>>>(motion, buf_d, out,
                                                                 out + MROWS * DMOT);
}
